// LocalActivationUnit_57999238365384
// MI455X (gfx1250) — matrix operand order repaired and checked by reading, not compiled, not run on hardware
//
#include <hip/hip_runtime.h>
#include <hip/hip_bf16.h>

typedef __attribute__((ext_vector_type(16))) _Float16 v16h;
typedef __attribute__((ext_vector_type(8)))  _Float16 v8h;
typedef __attribute__((ext_vector_type(8)))  float    v8f;

#define TT    200
#define DD    64
#define HH1   64
#define HH2   32
#define NTILE 13   // ceil(200/16)

__global__ __launch_bounds__(256)
void din_att_kernel(const float* __restrict__ behavior,   // (B,T,D)
                    const float* __restrict__ candidate,  // (B,D)
                    const int*   __restrict__ mask,       // (B,T)
                    const float* __restrict__ W1,         // (256,64)
                    const float* __restrict__ b1,         // (64)
                    const float* __restrict__ a1,         // (1)
                    const float* __restrict__ W2,         // (64,32)
                    const float* __restrict__ b2,         // (32)
                    const float* __restrict__ a2,         // (1)
                    const float* __restrict__ W3,         // (32,1)
                    const float* __restrict__ b3,         // (1)
                    float*       __restrict__ out)        // (B,D)
{
    // Effective layer-1 weight (cand folded in), f16, column-major: [c*64 + k]
    __shared__ _Float16 sWeffT[DD * HH1]  __attribute__((aligned(32)));
    __shared__ _Float16 sW2T[HH1 * HH2]   __attribute__((aligned(32))); // [c*64 + k]
    __shared__ _Float16 sH1[8][16 * HH1]  __attribute__((aligned(32))); // per-wave h1 tile
    __shared__ float sB1[HH1];  // effective bias: b1 + cand@(W1b - W1c)
    __shared__ float sB2[HH2], sW3v[HH2];
    __shared__ float sLogits[NTILE * 16];
    __shared__ float sRed[256];
    __shared__ float sScal[3]; // a1, a2, b3

    const int tid  = threadIdx.x;
    const int b    = blockIdx.x;
    const int lane = tid & 31;
    const int wave = tid >> 5;
    const int mrow = lane & 15;  // A-matrix row / B,C column selector
    const int kgrp = lane >> 4;  // K half selector (lanes 0-15 vs 16-31)

    const float* candRow = candidate + (size_t)b * DD;

    // ---------------- setup ----------------
    // Weff[k,c] = W1a[k,c] + W1c[k,c] + cand[k]*W1d[k,c]   (rows: a=0.., b=64.., c=128.., d=192..)
    for (int i = tid; i < DD * HH1; i += 256) {
        const int k = i >> 6, c = i & 63;
        const float ck = candRow[k];
        const float w  = W1[(size_t)k * HH1 + c]
                       + W1[(size_t)(128 + k) * HH1 + c]
                       + ck * W1[(size_t)(192 + k) * HH1 + c];
        sWeffT[c * DD + k] = (_Float16)w;
    }
    for (int i = tid; i < HH1 * HH2; i += 256) {
        const int k = i >> 5, c = i & 31;            // W2 row-major (k, c)
        sW2T[c * HH1 + k] = (_Float16)W2[i];
    }
    if (tid < HH1) { // beff[c] = b1[c] + sum_k cand[k]*(W1b[k,c] - W1c[k,c])
        float s = b1[tid];
        for (int k = 0; k < DD; ++k)
            s += candRow[k] * (W1[(size_t)(64 + k) * HH1 + tid]
                             - W1[(size_t)(128 + k) * HH1 + tid]);
        sB1[tid] = s;
    }
    if (tid < HH2) sB2[tid]  = b2[tid];
    if (tid < HH2) sW3v[tid] = W3[tid];
    if (tid == 0) { sScal[0] = a1[0]; sScal[1] = a2[0]; sScal[2] = b3[0]; }
    __syncthreads();

    const float va1 = sScal[0], va2 = sScal[1], vb3 = sScal[2];

    // ---------------- phase 1: per-wave MLP over 16-row T tiles ----------------
    for (int tile = wave; tile < NTILE; tile += 8) {
        const int  t0    = tile * 16;
        const int  r     = t0 + mrow;
        const bool valid = (r < TT);
        // Clamp to a legal row: garbage rows only pollute their own M-row, and
        // the logit write below is guarded, so loads stay unconditional (no exec branching).
        const float* behRow = behavior + ((size_t)b * TT + (valid ? r : 0)) * DD;
        __builtin_prefetch(behRow, 0, 1); // global_prefetch_b8

        // ---- layer 1: (16x64) @ (64x64) [Weff], 2 K-steps of 32 ----
        v8f acc[4] = {v8f{}, v8f{}, v8f{}, v8f{}};
#pragma unroll
        for (int kb = 0; kb < 2; ++kb) {
            // A in contiguous k order (same as B): lanes0-15 h[0..15]=K0..15 (+16 for lanes16-31)
            v16h afrag;
#pragma unroll
            for (int half = 0; half < 2; ++half) {
                const int f0 = kb * 32 + kgrp * 16 + half * 8;
                const float4 lo = *(const float4*)&behRow[f0];
                const float4 hi = *(const float4*)&behRow[f0 + 4];
                afrag[half * 8 + 0] = (_Float16)lo.x;
                afrag[half * 8 + 1] = (_Float16)lo.y;
                afrag[half * 8 + 2] = (_Float16)lo.z;
                afrag[half * 8 + 3] = (_Float16)lo.w;
                afrag[half * 8 + 4] = (_Float16)hi.x;
                afrag[half * 8 + 5] = (_Float16)hi.y;
                afrag[half * 8 + 6] = (_Float16)hi.z;
                afrag[half * 8 + 7] = (_Float16)hi.w;
            }
#pragma unroll
            for (int nt = 0; nt < 4; ++nt) {
                const int col = nt * 16 + mrow;
                const v16h bfrag = *(const v16h*)&sWeffT[col * DD + kb * 32 + kgrp * 16];
                acc[nt] = __builtin_amdgcn_wmma_f32_16x16x32_f16(
                    false, afrag, false, bfrag, (short)0, acc[nt], false, false);
            }
        }

        // ---- effective bias + PReLU, spill h1 tile to LDS (C-layout -> A-layout) ----
#pragma unroll
        for (int nt = 0; nt < 4; ++nt) {
            const int   col = nt * 16 + mrow;
            const float bb  = sB1[col];
#pragma unroll
            for (int j = 0; j < 8; ++j) {
                const int m = kgrp * 8 + j;          // lanes16-31 hold M=8..15
                float x = acc[nt][j] + bb;
                x = (x >= 0.0f) ? x : va1 * x;
                sH1[wave][m * HH1 + col] = (_Float16)x;
            }
        }
        asm volatile("s_wait_dscnt 0x0" ::: "memory"); // wave-internal LDS RAW fence

        // ---- layer 2: (16x64) @ (64x32), 2 K-steps ----
        v8f acc2[2] = {v8f{}, v8f{}};
#pragma unroll
        for (int kb = 0; kb < 2; ++kb) {
            const v8h lo = *(const v8h*)&sH1[wave][mrow * HH1 + kb * 32 + kgrp * 16];
            const v8h hi = *(const v8h*)&sH1[wave][mrow * HH1 + kb * 32 + kgrp * 16 + 8];
            v16h afrag;
#pragma unroll
            for (int i = 0; i < 8; ++i) { afrag[i] = lo[i]; afrag[8 + i] = hi[i]; }
#pragma unroll
            for (int nt = 0; nt < 2; ++nt) {
                const int col = nt * 16 + mrow;
                const v16h bfrag = *(const v16h*)&sW2T[col * HH1 + kb * 32 + kgrp * 16];
                acc2[nt] = __builtin_amdgcn_wmma_f32_16x16x32_f16(
                    false, afrag, false, bfrag, (short)0, acc2[nt], false, false);
            }
        }

        // ---- layer 3: (16x32) @ (32x1) via VALU + 16-lane butterfly ----
        const float w3a = sW3v[mrow], w3b = sW3v[16 + mrow];
        const float b2a = sB2[mrow],  b2b = sB2[16 + mrow];
        float p[8];
#pragma unroll
        for (int j = 0; j < 8; ++j) {
            float x0 = acc2[0][j] + b2a; x0 = (x0 >= 0.0f) ? x0 : va2 * x0;
            float x1 = acc2[1][j] + b2b; x1 = (x1 >= 0.0f) ? x1 : va2 * x1;
            p[j] = x0 * w3a + x1 * w3b;
        }
#pragma unroll
        for (int off = 8; off >= 1; off >>= 1) {
#pragma unroll
            for (int j = 0; j < 8; ++j) p[j] += __shfl_xor(p[j], off, 16);
        }
        if (mrow == 0) { // lane 0 -> rows t0..t0+7, lane 16 -> rows t0+8..t0+15
#pragma unroll
            for (int j = 0; j < 8; ++j) {
                const int rr = t0 + kgrp * 8 + j;
                if (rr < TT) {
                    const float lg = p[j] + vb3;
                    sLogits[rr] = mask[(size_t)b * TT + rr] ? lg : -__builtin_inff();
                }
            }
        }
    }
    __syncthreads();

    // ---------------- phase 2: masked softmax over T ----------------
    float lm = -__builtin_inff();
    if (tid < TT) lm = sLogits[tid];
    sRed[tid] = lm;
    __syncthreads();
    for (int s = 128; s > 0; s >>= 1) {
        if (tid < s) sRed[tid] = fmaxf(sRed[tid], sRed[tid + s]);
        __syncthreads();
    }
    const float mx = sRed[0];
    __syncthreads();
    float ev = 0.0f;
    if (tid < TT) { ev = __expf(lm - mx); sLogits[tid] = ev; }
    sRed[tid] = ev;
    __syncthreads();
    for (int s = 128; s > 0; s >>= 1) {
        if (tid < s) sRed[tid] += sRed[tid + s];
        __syncthreads();
    }
    const float inv = 1.0f / sRed[0];
    __syncthreads();

    // ---------------- phase 3: out[b,d] = sum_t w[t] * behavior[b,t,d] ----------------
    const int d   = tid & 63;
    const int grp = tid >> 6; // 4 T-slices
    float accum = 0.0f;
    for (int t = grp; t < TT; t += 4) {
        accum += sLogits[t] * behavior[((size_t)b * TT + t) * DD + d];
    }
    sRed[tid] = accum;
    __syncthreads();
    if (tid < 64) {
        const float s = sRed[d] + sRed[64 + d] + sRed[128 + d] + sRed[192 + d];
        out[(size_t)b * DD + d] = s * inv;
    }
}

extern "C" void kernel_launch(void* const* d_in, const int* in_sizes, int n_in,
                              void* d_out, int out_size, void* d_ws, size_t ws_size,
                              hipStream_t stream) {
    (void)n_in; (void)out_size; (void)d_ws; (void)ws_size;
    const float* behavior  = (const float*)d_in[0];
    const float* candidate = (const float*)d_in[1];
    const int*   mask      = (const int*)  d_in[2];
    const float* W1        = (const float*)d_in[3];
    const float* b1        = (const float*)d_in[4];
    const float* a1        = (const float*)d_in[5];
    const float* W2        = (const float*)d_in[6];
    const float* b2        = (const float*)d_in[7];
    const float* a2        = (const float*)d_in[8];
    const float* W3        = (const float*)d_in[9];
    const float* b3        = (const float*)d_in[10];
    float*       out       = (float*)d_out;

    const int Bn = in_sizes[1] / DD; // candidate is (B, D)
    din_att_kernel<<<dim3(Bn), dim3(256), 0, stream>>>(
        behavior, candidate, mask, W1, b1, a1, W2, b2, a2, W3, b3, out);
}
